// DualGridMLMMelHarm_18399639896308
// MI455X (gfx1250) — matrix operand order repaired and checked by reading, not compiled, not run on hardware
//
#include <hip/hip_runtime.h>
#include <hip/hip_bf16.h>
#include <math.h>

typedef __attribute__((ext_vector_type(16))) _Float16 v16h;
typedef __attribute__((ext_vector_type(8)))  _Float16 v8h;
typedef __attribute__((ext_vector_type(4)))  _Float16 v4h;
typedef __attribute__((ext_vector_type(8)))  float    v8f;
typedef unsigned int v4u __attribute__((ext_vector_type(4)));
typedef int          v8i __attribute__((ext_vector_type(8)));
typedef int          v4i __attribute__((ext_vector_type(4)));

namespace {

constexpr int kB    = 32;
constexpr int kL    = 256;   // LM == LH
constexpr int kD    = 512;
constexpr int kNH   = 8;
constexpr int kDH   = 64;
constexpr int kDFF  = 2048;
constexpr int kV    = 512;
constexpr int kPR   = 13;
constexpr int kRows = kB * kL; // 8192

__device__ __forceinline__ float pe_val(int l, int d) {
  int i = d >> 1;
  float freq = expf((float)(2 * i) * (-9.210340371976184f / (float)kD)); // -ln(10000)/D
  float ang = (float)l * freq;
  return (d & 1) ? cosf(ang) : sinf(ang);
}

// ---------------------------------------------------------------------------
// TDM: DMA a 2-D f16 tile (tile_rows x 32 elems) global -> LDS.
// LDS layout produced: rows padded to 56 halves (112B) via D# pad fields:
//   pad_interval=3 (pad after every 16 DWORDs = one 64B row),
//   pad_amount=11  (12 DWORDs = 48B)  => LDS row stride = 112B.
// tensor_dim1 is set huge so no OOB zeroing occurs (tiles are always in-range).
// ---------------------------------------------------------------------------
__device__ __forceinline__ void tdm_load_tile(unsigned lds_addr, const void* gaddr,
                                              unsigned tile_rows, unsigned ld_elems) {
  const unsigned long long ga = (unsigned long long)gaddr;
  v4u g0;
  g0.x = 1u;                                   // count=1 (valid), no gather
  g0.y = lds_addr;                             // LDS byte address
  g0.z = (unsigned)ga;                         // global_addr[31:0]
  g0.w = (unsigned)((ga >> 32) & 0x01FFFFFFu)  // global_addr[56:32]
         | 0x80000000u;                        // type=2 ("image")
  const unsigned td1 = 0x40000000u;            // huge tensor_dim1
  v8i g1;
  g1[0] = (int)((1u << 16)        // data_size = 2 bytes
              | (1u << 20)        // pad_enable
              | (3u << 22)        // pad_interval: 16 DWORDs
              | (11u << 25));     // pad_amount : 12 DWORDs
  g1[1] = (int)((ld_elems & 0xFFFFu) << 16);                       // tensor_dim0 lo
  g1[2] = (int)(((ld_elems >> 16) & 0xFFFFu) | ((td1 & 0xFFFFu) << 16));
  g1[3] = (int)(((td1 >> 16) & 0xFFFFu) | (32u << 16));            // tile_dim0 = 32
  g1[4] = (int)(tile_rows & 0xFFFFu);                              // tile_dim1; tile_dim2=0
  g1[5] = (int)ld_elems;                                           // dim0_stride lo32
  g1[6] = 0;                                                       // stride hi16, d1stride lo
  g1[7] = 0;
  const v4i z4 = {0, 0, 0, 0};
  const v8i z8 = {0, 0, 0, 0, 0, 0, 0, 0};
  // 6-arg form (this toolchain): (g0, g1, g2, g3, g4, cpol)
  __builtin_amdgcn_tensor_load_to_lds(g0, g1, z4, z4, z8, 0);
}

__device__ __forceinline__ unsigned lds_off(const void* p) {
  return (unsigned)(unsigned long long)p;  // low 32 bits of generic ptr = LDS offset
}

// ---------------------------------------------------------------------------
// TDM-staged, double-buffered WMMA GEMM. B is (N,K) row-major weights.
//   C = act( scale * A @ B^T + bias ) + R -> C32 (f32) and/or C16 (f16)
// Block: 256 thr = 8 waves (MW x NW); wave tile 64x32 (4x2 WMMA 16x16x32).
// Wave 0 drives the TDM engine; TENSORcnt + split barriers for sync.
// ---------------------------------------------------------------------------
template<int MW, int NW, bool GELU>
__global__ __launch_bounds__(256)
void gemm16_tdm(const _Float16* __restrict__ A, const _Float16* __restrict__ B,
                float* __restrict__ C32, _Float16* __restrict__ C16,
                const float* __restrict__ bias, const float* __restrict__ R,
                int K, int lda, int ldb, int ldc, int nh,
                long sAb, long sAh, long sBb, long sBh, long sCb, long sCh,
                float scale)
{
  constexpr int MH  = MW * 64;
  constexpr int NB  = NW * 32;
  constexpr int LDS = 56;      // padded halves per row (112B, conflict-free)
  __shared__ _Float16 As[2][MH * LDS];
  __shared__ _Float16 Bs[2][NB * LDS];

  const int tid    = threadIdx.x;
  const int lane   = tid & 31;
  const int lane16 = lane & 15;
  const int half   = (lane >> 4) & 1;
  const int w      = tid >> 5;
  const int wm     = w / NW, wn = w % NW;

  const long g  = blockIdx.z;
  const long bb = g / nh, hh = g % nh;
  const _Float16* Ab = A + bb * sAb + hh * sAh;
  const _Float16* Bb = B + bb * sBb + hh * sBh;
  const int m0 = blockIdx.x * MH;
  const int n0 = blockIdx.y * NB;

  v8f acc[4][2] = {};
  const int nk = K / 32;

  if (w == 0) {
    tdm_load_tile(lds_off(&As[0][0]), Ab + (long)m0 * lda, MH, (unsigned)lda);
    tdm_load_tile(lds_off(&Bs[0][0]), Bb + (long)n0 * ldb, NB, (unsigned)ldb);
  }
  for (int ik = 0; ik < nk; ++ik) {
    const int cur = ik & 1;
    if (w == 0) {
      if (ik + 1 < nk) {
        const int kk = (ik + 1) * 32;
        tdm_load_tile(lds_off(&As[cur ^ 1][0]), Ab + (long)m0 * lda + kk, MH, (unsigned)lda);
        tdm_load_tile(lds_off(&Bs[cur ^ 1][0]), Bb + (long)n0 * ldb + kk, NB, (unsigned)ldb);
        __builtin_amdgcn_s_wait_tensorcnt(2);  // current pair complete (in-order)
      } else {
        __builtin_amdgcn_s_wait_tensorcnt(0);
      }
    }
    __syncthreads();  // publish LDS tile `cur`
    v16h bf[2];
#pragma unroll
    for (int nt = 0; nt < 2; ++nt) {
      const int n = wn * 32 + nt * 16 + lane16;
      union { v16h f; v8h h[2]; } u;
      u.h[0] = *(const v8h*)(&Bs[cur][0] + n * LDS + half * 16);
      u.h[1] = *(const v8h*)(&Bs[cur][0] + n * LDS + half * 16 + 8);
      bf[nt] = u.f;
    }
#pragma unroll
    for (int mt = 0; mt < 4; ++mt) {
      const int m = wm * 64 + mt * 16 + lane16;
      union { v16h f; v8h h[2]; } u;
      u.h[0] = *(const v8h*)(&As[cur][0] + m * LDS + half * 16);
      u.h[1] = *(const v8h*)(&As[cur][0] + m * LDS + half * 16 + 8);
      acc[mt][0] = __builtin_amdgcn_wmma_f32_16x16x32_f16(
          false, u.f, false, bf[0], (short)0, acc[mt][0], false, false);
      acc[mt][1] = __builtin_amdgcn_wmma_f32_16x16x32_f16(
          false, u.f, false, bf[1], (short)0, acc[mt][1], false, false);
    }
    __syncthreads();  // everyone done reading `cur` before TDM overwrites it
  }

  float*     C32b = C32 ? C32 + bb * sCb + hh * sCh : nullptr;
  _Float16*  C16b = C16 ? C16 + bb * sCb + hh * sCh : nullptr;
  const float* Rb = R   ? R   + bb * sCb + hh * sCh : nullptr;
#pragma unroll
  for (int mt = 0; mt < 4; ++mt) {
#pragma unroll
    for (int nt = 0; nt < 2; ++nt) {
      const int n = n0 + wn * 32 + nt * 16 + lane16;
      const float bn = bias ? bias[n] : 0.0f;
#pragma unroll
      for (int r = 0; r < 8; ++r) {
        const int m = m0 + wm * 64 + mt * 16 + r + half * 8;
        float v = acc[mt][nt][r] * scale + bn;
        if (GELU) v = 0.5f * v * (1.0f + erff(v * 0.70710678118654752f));
        if (Rb) v += Rb[(long)m * ldc + n];
        if (C32b) C32b[(long)m * ldc + n] = v;
        if (C16b) C16b[(long)m * ldc + n] = (_Float16)v;
      }
    }
  }
}

// ---------------------------------------------------------------------------
// Manual-staging WMMA GEMM for B given as (K,N) row-major (P @ V):
// B is transposed into LDS during staging. Single-buffered.
// ---------------------------------------------------------------------------
template<int MW, int NW>
__global__ __launch_bounds__(256)
void gemm16_kn(const _Float16* __restrict__ A, const _Float16* __restrict__ B,
               _Float16* __restrict__ C16,
               int K, int lda, int ldb, int ldc, int nh,
               long sAb, long sAh, long sBb, long sBh, long sCb, long sCh)
{
  constexpr int MH  = MW * 64;
  constexpr int NB  = NW * 32;
  constexpr int LDS = 56;
  __shared__ _Float16 As[MH * LDS];
  __shared__ _Float16 Bs[NB * LDS];

  const int tid    = threadIdx.x;
  const int lane   = tid & 31;
  const int lane16 = lane & 15;
  const int half   = (lane >> 4) & 1;
  const int w      = tid >> 5;
  const int wm     = w / NW, wn = w % NW;

  const long g  = blockIdx.z;
  const long bb = g / nh, hh = g % nh;
  const _Float16* Ab = A + bb * sAb + hh * sAh;
  const _Float16* Bb = B + bb * sBb + hh * sBh;
  const int m0 = blockIdx.x * MH;
  const int n0 = blockIdx.y * NB;

  v8f acc[4][2] = {};

  for (int kk = 0; kk < K; kk += 32) {
    __syncthreads();
#pragma unroll
    for (int it = 0; it < (MH * 4) / 256; ++it) {
      const int idx = tid + it * 256;
      const int r = idx >> 2, c = idx & 3;
      const v8h val = *(const v8h*)(Ab + (long)(m0 + r) * lda + kk + c * 8);
      *(v8h*)(As + r * LDS + c * 8) = val;
    }
#pragma unroll
    for (int it = 0; it < (32 * (NB / 8)) / 256; ++it) {
      const int idx = tid + it * 256;
      const int kr = idx / (NB / 8), c = idx % (NB / 8);
      const v8h val = *(const v8h*)(Bb + (long)(kk + kr) * ldb + n0 + c * 8);
#pragma unroll
      for (int j = 0; j < 8; ++j)
        Bs[(c * 8 + j) * LDS + kr] = val[j];
    }
    __syncthreads();
    v16h bf[2];
#pragma unroll
    for (int nt = 0; nt < 2; ++nt) {
      const int n = wn * 32 + nt * 16 + lane16;
      union { v16h f; v8h h[2]; } u;
      u.h[0] = *(const v8h*)(Bs + n * LDS + half * 16);
      u.h[1] = *(const v8h*)(Bs + n * LDS + half * 16 + 8);
      bf[nt] = u.f;
    }
#pragma unroll
    for (int mt = 0; mt < 4; ++mt) {
      const int m = wm * 64 + mt * 16 + lane16;
      union { v16h f; v8h h[2]; } u;
      u.h[0] = *(const v8h*)(As + m * LDS + half * 16);
      u.h[1] = *(const v8h*)(As + m * LDS + half * 16 + 8);
      acc[mt][0] = __builtin_amdgcn_wmma_f32_16x16x32_f16(
          false, u.f, false, bf[0], (short)0, acc[mt][0], false, false);
      acc[mt][1] = __builtin_amdgcn_wmma_f32_16x16x32_f16(
          false, u.f, false, bf[1], (short)0, acc[mt][1], false, false);
    }
  }

  _Float16* C16b = C16 + bb * sCb + hh * sCh;
#pragma unroll
  for (int mt = 0; mt < 4; ++mt) {
#pragma unroll
    for (int nt = 0; nt < 2; ++nt) {
      const int n = n0 + wn * 32 + nt * 16 + lane16;
#pragma unroll
      for (int r = 0; r < 8; ++r) {
        const int m = m0 + wm * 64 + mt * 16 + r + half * 8;
        C16b[(long)m * ldc + n] = (_Float16)acc[mt][nt][r];
      }
    }
  }
}

// Row softmax over 256 columns: f32 in, f16 probabilities out.
__global__ __launch_bounds__(256)
void softmax_rows(const float* __restrict__ S, _Float16* __restrict__ P) {
  __shared__ float red[256];
  const float* p = S + (long)blockIdx.x * 256;
  _Float16*    q = P + (long)blockIdx.x * 256;
  const int t = threadIdx.x;
  float v = p[t];
  red[t] = v; __syncthreads();
  for (int s = 128; s > 0; s >>= 1) { if (t < s) red[t] = fmaxf(red[t], red[t + s]); __syncthreads(); }
  const float mx = red[0];
  __syncthreads();
  const float e = expf(v - mx);
  red[t] = e; __syncthreads();
  for (int s = 128; s > 0; s >>= 1) { if (t < s) red[t] += red[t + s]; __syncthreads(); }
  q[t] = (_Float16)(e / red[0]);
}

// LayerNorm(X (+H)) * g + b over D=512; dual f32/f16 output. In-place safe.
__global__ __launch_bounds__(256)
void layernorm_rows(const float* __restrict__ X, const float* __restrict__ H,
                    float* __restrict__ Y32, _Float16* __restrict__ Y16,
                    const float* __restrict__ gw, const float* __restrict__ bw) {
  __shared__ float red[256];
  const long base = (long)blockIdx.x * kD;
  const int t = threadIdx.x;
  float v0 = X[base + t];
  float v1 = X[base + t + 256];
  if (H) { v0 += H[base + t]; v1 += H[base + t + 256]; }
  red[t] = v0 + v1; __syncthreads();
  for (int s = 128; s > 0; s >>= 1) { if (t < s) red[t] += red[t + s]; __syncthreads(); }
  const float mean = red[0] * (1.0f / kD);
  __syncthreads();
  const float d0 = v0 - mean, d1 = v1 - mean;
  red[t] = d0 * d0 + d1 * d1; __syncthreads();
  for (int s = 128; s > 0; s >>= 1) { if (t < s) red[t] += red[t + s]; __syncthreads(); }
  const float inv = rsqrtf(red[0] * (1.0f / kD) + 1e-5f);
  const float y0 = d0 * inv * gw[t]       + bw[t];
  const float y1 = d1 * inv * gw[t + 256] + bw[t + 256];
  Y32[base + t]       = y0;
  Y32[base + t + 256] = y1;
  Y16[base + t]       = (_Float16)y0;
  Y16[base + t + 256] = (_Float16)y1;
}

// out[row,:] = melody_grid[row,:] @ W^T + b + PE[row % L, :]
__global__ __launch_bounds__(256)
void mel_init(const float* __restrict__ grid, const float* __restrict__ W,
              const float* __restrict__ bias, float* __restrict__ out) {
  const long row = blockIdx.x;
  const int t = threadIdx.x;
  const int l = (int)(row % kL);
  float xr[kPR];
#pragma unroll
  for (int p = 0; p < kPR; ++p) xr[p] = grid[row * kPR + p];
#pragma unroll
  for (int j = 0; j < 2; ++j) {
    const int d = t + j * 256;
    float s = bias[d];
#pragma unroll
    for (int p = 0; p < kPR; ++p) s += xr[p] * W[(long)d * kPR + p];
    out[row * kD + d] = s + pe_val(l, d);
  }
}

// out[row,:] = emb[tok[row],:] + PE[row % L, :]
__global__ __launch_bounds__(256)
void harm_init(const int* __restrict__ tok, const float* __restrict__ emb,
               float* __restrict__ out) {
  const long row = blockIdx.x;
  const int t = threadIdx.x;
  const int l = (int)(row % kL);
  const long token = tok[row];
  for (int j = 0; j < 2; ++j) {
    const int d = t + j * 256;
    out[row * kD + d] = emb[token * kD + d] + pe_val(l, d);
  }
}

// f32 -> f16 cast, 4 elements per thread (n must be a multiple of 4).
__global__ __launch_bounds__(256)
void cast_f32_f16(const float* __restrict__ X, _Float16* __restrict__ Y, long n) {
  const long i = ((long)blockIdx.x * 256 + threadIdx.x) * 4;
  if (i < n) {
    const float4 v = *(const float4*)(X + i);
    v4h o; o[0] = (_Float16)v.x; o[1] = (_Float16)v.y;
           o[2] = (_Float16)v.z; o[3] = (_Float16)v.w;
    *(v4h*)(Y + i) = o;
  }
}

static inline void cast16(hipStream_t st, const float* src, _Float16* dst, long n) {
  cast_f32_f16<<<dim3((unsigned)((n / 4 + 255) / 256)), dim3(256), 0, st>>>(src, dst, n);
}

// Standard GEMM: block tile 128x128 (waves 2x4), TDM-staged, B is (N,K).
static inline void g_std(hipStream_t st, bool gelu,
    const _Float16* A, const _Float16* B, float* C32, _Float16* C16,
    const float* bias, const float* R,
    int M, int N, int K, int lda, int ldb, int ldc,
    int batch, int nh, long sAb, long sAh, long sBb, long sBh, long sCb, long sCh,
    float scale)
{
  dim3 grid(M / 128, N / 128, batch), blk(256);
  if (gelu)
    gemm16_tdm<2, 4, true><<<grid, blk, 0, st>>>(A, B, C32, C16, bias, R, K,
        lda, ldb, ldc, nh, sAb, sAh, sBb, sBh, sCb, sCh, scale);
  else
    gemm16_tdm<2, 4, false><<<grid, blk, 0, st>>>(A, B, C32, C16, bias, R, K,
        lda, ldb, ldc, nh, sAb, sAh, sBb, sBh, sCb, sCh, scale);
}

// P@V GEMM: block tile 256x64 (waves 4x2), B is (K,N) row-major.
static inline void g_pv(hipStream_t st,
    const _Float16* A, const _Float16* B, _Float16* C16,
    int M, int N, int K, int lda, int ldb, int ldc,
    int batch, int nh, long sAb, long sAh, long sBb, long sBh, long sCb, long sCh)
{
  dim3 grid(M / 256, N / 64, batch), blk(256);
  gemm16_kn<4, 2><<<grid, blk, 0, st>>>(A, B, C16, K, lda, ldb, ldc, nh,
      sAb, sAh, sBb, sBh, sCb, sCh);
}

} // namespace

extern "C" void kernel_launch(void* const* d_in, const int* in_sizes, int n_in,
                              void* d_out, int out_size, void* d_ws, size_t ws_size,
                              hipStream_t stream) {
  (void)in_sizes; (void)n_in; (void)out_size; (void)ws_size;
  auto F = [&](int i) { return (const float*)d_in[i]; };
  // setup_inputs() leaf order (dict insertion order):
  // 0 melody_grid  1 harmony_tokens  2 mel_proj.w  3 mel_proj.b  4 harm_emb
  // mel_layers:  5 qkv_w 6 qkv_b 7 out_w 8 out_b 9 l1_w 10 l1_b 11 l2_w 12 l2_b
  //              13 ln1_g 14 ln1_b 15 ln2_g 16 ln2_b
  // harm_layers: 17 qkv_w 18 qkv_b 19 out_w 20 out_b 21 l1_w 22 l1_b 23 l2_w 24 l2_b
  //              25 ln1_g 26 ln1_b 27 ln2_g 28 ln2_b 29 cqkv_w 30 cqkv_b 31 cout_w
  //              32 cout_b 33 ln3_g 34 ln3_b
  // norms: 35 in_mel_g 36 in_mel_b 37 out_mel_g 38 out_mel_b
  //        39 in_harm_g 40 in_harm_b 41 out_harm_g 42 out_harm_b
  // 43 out_head.w  44 out_head.b
  const float* melody_grid = F(0);
  const int*   harm_tok    = (const int*)d_in[1];

  // ---- workspace layout ----
  char* ws = (char*)d_ws;
  float*    x_mel32  = (float*)   (ws + 0);                  // 16 MB (8192x512 f32)
  float*    x_harm32 = (float*)   (ws + (size_t)16777216);   // 16 MB
  float*    tbuf32   = (float*)   (ws + (size_t)33554432);   // 16 MB
  float*    sc32     = (float*)   (ws + (size_t)50331648);   // 64 MB scores f32
  _Float16* qkv16    = (_Float16*)(ws + (size_t)117440512);  // 24 MB (8192x1536 f16)
  _Float16* p16      = (_Float16*)(ws + (size_t)142606336);  // 32 MB probs f16 / ffn-hid f16
  _Float16* ctx16    = (_Float16*)(ws + (size_t)176160768);  //  8 MB (8192x512 f16)
  _Float16* mel16    = (_Float16*)(ws + (size_t)184549376);  //  8 MB melody activations f16
  _Float16* x16      = (_Float16*)(ws + (size_t)192937984);  //  8 MB harmony LN outputs f16
  _Float16* wl16     = (_Float16*)(ws + (size_t)201326592);  //  8 MB per-layer f16 weights
  _Float16* w_qkv  = wl16;
  _Float16* w_out  = wl16 + 786432;
  _Float16* w_l1   = wl16 + 1048576;
  _Float16* w_l2   = wl16 + 2097152;
  _Float16* w_cqkv = wl16 + 3145728;
  _Float16* w_cout = wl16 + 3932160;

  const long sQb   = (long)kL * 3 * kD;    // qkv per-batch stride (elements)
  const long sScb  = (long)kNH * kL * kL;  // scores/probs per-batch stride
  const long sScH  = (long)kL * kL;        // per-head stride
  const long sCtxb = (long)kL * kD;        // ctx per-batch stride

  // attention on qkv16 (8192x1536): q cols [0,512), k [512,1024), v [1024,1536)
  auto attention = [&]() {
    g_std(stream, false, qkv16, qkv16 + kD, sc32, nullptr, nullptr, nullptr,
          kL, kL, kDH, 3 * kD, 3 * kD, kL, kB * kNH, kNH,
          sQb, kDH, sQb, kDH, sScb, sScH, 0.125f);
    softmax_rows<<<dim3(kB * kNH * kL), dim3(256), 0, stream>>>(sc32, p16);
    g_pv(stream, p16, qkv16 + 2 * kD, ctx16,
         kL, kDH, kL, kL, 3 * kD, kD, kB * kNH, kNH,
         sScb, sScH, sQb, kDH, sCtxb, kDH);
  };

  // ================= melody encoder (post-norm) =================
  mel_init<<<dim3(kRows), dim3(256), 0, stream>>>(melody_grid, F(2), F(3), x_mel32);
  layernorm_rows<<<dim3(kRows), dim3(256), 0, stream>>>(x_mel32, nullptr, x_mel32, mel16, F(35), F(36));
  for (int l = 0; l < 8; ++l) {
    cast16(stream, F(5)  + (long)l * 3 * kD * kD,  w_qkv, (long)3 * kD * kD);
    cast16(stream, F(7)  + (long)l * kD * kD,      w_out, (long)kD * kD);
    cast16(stream, F(9)  + (long)l * kDFF * kD,    w_l1,  (long)kDFF * kD);
    cast16(stream, F(11) + (long)l * kD * kDFF,    w_l2,  (long)kD * kDFF);
    const float* qkv_b = F(6)  + (long)l * 3 * kD;
    const float* out_b = F(8)  + (long)l * kD;
    const float* l1_b  = F(10) + (long)l * kDFF;
    const float* l2_b  = F(12) + (long)l * kD;
    g_std(stream, false, mel16, w_qkv, nullptr, qkv16, qkv_b, nullptr,
          kRows, 3 * kD, kD, kD, kD, 3 * kD, 1, 1, 0, 0, 0, 0, 0, 0, 1.0f);
    attention();
    g_std(stream, false, ctx16, w_out, tbuf32, nullptr, out_b, nullptr,
          kRows, kD, kD, kD, kD, kD, 1, 1, 0, 0, 0, 0, 0, 0, 1.0f);
    layernorm_rows<<<dim3(kRows), dim3(256), 0, stream>>>(x_mel32, tbuf32, x_mel32, mel16,
        F(13) + (long)l * kD, F(14) + (long)l * kD);
    g_std(stream, true, mel16, w_l1, nullptr, p16, l1_b, nullptr,
          kRows, kDFF, kD, kD, kD, kDFF, 1, 1, 0, 0, 0, 0, 0, 0, 1.0f);
    g_std(stream, false, p16, w_l2, tbuf32, nullptr, l2_b, nullptr,
          kRows, kD, kDFF, kDFF, kDFF, kD, 1, 1, 0, 0, 0, 0, 0, 0, 1.0f);
    layernorm_rows<<<dim3(kRows), dim3(256), 0, stream>>>(x_mel32, tbuf32, x_mel32, mel16,
        F(15) + (long)l * kD, F(16) + (long)l * kD);
  }
  layernorm_rows<<<dim3(kRows), dim3(256), 0, stream>>>(x_mel32, nullptr, x_mel32, mel16, F(37), F(38));

  // ================= harmony encoder (pre-norm + cross) =================
  harm_init<<<dim3(kRows), dim3(256), 0, stream>>>(harm_tok, F(4), x_harm32);
  layernorm_rows<<<dim3(kRows), dim3(256), 0, stream>>>(x_harm32, nullptr, x_harm32, x16, F(39), F(40));
  for (int l = 0; l < 8; ++l) {
    cast16(stream, F(17) + (long)l * 3 * kD * kD,  w_qkv,  (long)3 * kD * kD);
    cast16(stream, F(19) + (long)l * kD * kD,      w_out,  (long)kD * kD);
    cast16(stream, F(21) + (long)l * kDFF * kD,    w_l1,   (long)kDFF * kD);
    cast16(stream, F(23) + (long)l * kD * kDFF,    w_l2,   (long)kD * kDFF);
    cast16(stream, F(29) + (long)l * 3 * kD * kD,  w_cqkv, (long)3 * kD * kD);
    cast16(stream, F(31) + (long)l * kD * kD,      w_cout, (long)kD * kD);
    const float* qkv_b  = F(18) + (long)l * 3 * kD;
    const float* out_b  = F(20) + (long)l * kD;
    const float* l1_b   = F(22) + (long)l * kDFF;
    const float* l2_b   = F(24) + (long)l * kD;
    const float* cqkv_b = F(30) + (long)l * 3 * kD;
    const float* cout_b = F(32) + (long)l * kD;

    layernorm_rows<<<dim3(kRows), dim3(256), 0, stream>>>(x_harm32, nullptr, tbuf32, x16,
        F(25) + (long)l * kD, F(26) + (long)l * kD);
    g_std(stream, false, x16, w_qkv, nullptr, qkv16, qkv_b, nullptr,
          kRows, 3 * kD, kD, kD, kD, 3 * kD, 1, 1, 0, 0, 0, 0, 0, 0, 1.0f);
    attention();
    g_std(stream, false, ctx16, w_out, x_harm32, nullptr, out_b, tbuf32,
          kRows, kD, kD, kD, kD, kD, 1, 1, 0, 0, 0, 0, 0, 0, 1.0f);
    layernorm_rows<<<dim3(kRows), dim3(256), 0, stream>>>(x_harm32, nullptr, tbuf32, x16,
        F(27) + (long)l * kD, F(28) + (long)l * kD);
    g_std(stream, false, x16, w_cqkv, nullptr, qkv16, cqkv_b, nullptr,
          kRows, kD, kD, kD, kD, 3 * kD, 1, 1, 0, 0, 0, 0, 0, 0, 1.0f);
    g_std(stream, false, mel16, w_cqkv + (long)kD * kD, nullptr, qkv16 + kD,
          cqkv_b + kD, nullptr,
          kRows, 2 * kD, kD, kD, kD, 3 * kD, 1, 1, 0, 0, 0, 0, 0, 0, 1.0f);
    attention();
    g_std(stream, false, ctx16, w_cout, x_harm32, nullptr, cout_b, tbuf32,
          kRows, kD, kD, kD, kD, kD, 1, 1, 0, 0, 0, 0, 0, 0, 1.0f);
    layernorm_rows<<<dim3(kRows), dim3(256), 0, stream>>>(x_harm32, nullptr, tbuf32, x16,
        F(33) + (long)l * kD, F(34) + (long)l * kD);
    g_std(stream, true, x16, w_l1, nullptr, p16, l1_b, nullptr,
          kRows, kDFF, kD, kD, kD, kDFF, 1, 1, 0, 0, 0, 0, 0, 0, 1.0f);
    g_std(stream, false, p16, w_l2, x_harm32, nullptr, l2_b, tbuf32,
          kRows, kD, kDFF, kDFF, kDFF, kD, 1, 1, 0, 0, 0, 0, 0, 0, 1.0f);
  }
  layernorm_rows<<<dim3(kRows), dim3(256), 0, stream>>>(x_harm32, nullptr, x_harm32, x16, F(41), F(42));
  cast16(stream, F(43), w_out, (long)kV * kD);
  g_std(stream, false, x16, w_out, (float*)d_out, nullptr, F(44), nullptr,
        kRows, kV, kD, kD, kD, kV, 1, 1, 0, 0, 0, 0, 0, 0, 1.0f);
}
